// RoPEAttention_49091476193514
// MI455X (gfx1250) — matrix operand order repaired and checked by reading, not compiled, not run on hardware
//
#include <hip/hip_runtime.h>
#include <hip/hip_bf16.h>
#include <math.h>

typedef _Float16 f16;
typedef __attribute__((ext_vector_type(16))) _Float16 v16h;
typedef __attribute__((ext_vector_type(8)))  float    v8f;

#define HEADS   12
#define DHEAD   64
#define DIM     768
#define NTOK    577
#define BATCH   64
#define ROWS    (BATCH * NTOK)      /* 36928 = 577 * 64 */
#define QKVN    (3 * DIM)           /* 2304 */
#define NEG_BIG (-1e30f)

union U32H2 { unsigned u; f16 h[2]; };

// ---------------------------------------------------------------------------
// gfx1250 async global->LDS copy (ISA §15.18.3, GLOBAL_LOAD_ASYNC_TO_LDS_B128,
// tracked by ASYNCcnt). LDS byte address = low 32 bits of the generic pointer
// (flat->LDS aperture truncation, ISA §10.2).
// ---------------------------------------------------------------------------
__device__ __forceinline__ void async_copy_b128(void* lds_ptr, const void* gptr) {
  unsigned lds_off = (unsigned)(size_t)lds_ptr;
  asm volatile("global_load_async_to_lds_b128 %0, %1, off"
               :: "v"(lds_off), "v"(gptr) : "memory");
}
__device__ __forceinline__ void wait_async0() {
  asm volatile("s_wait_asynccnt 0x0" ::: "memory");
}

// ---------------------------------------------------------------------------
// WMMA A-fragment gather (16x32 f16, ISA §7.12.2): lane L holds row m = L&15;
// element idx holds k = idx + (L>=16 ? 16 : 0), this code's own k order (not the ISA table's; B matches)
// -> one contiguous 16-half run per lane (2x ds_load_b128).
// ---------------------------------------------------------------------------
__device__ __forceinline__ v16h load_a_frag(const f16* t, int ld, int mbase,
                                            int kbase, int lane) {
  v16h a;
  const int m   = mbase + (lane & 15);
  const int khi = (lane & 16) ? 16 : 0;
#pragma unroll
  for (int idx = 0; idx < 16; ++idx) {
    const int k  = idx + khi;
    a[idx] = t[m * ld + kbase + k];
  }
  return a;
}

// B-fragment from an LDS tile stored K-contiguous per column (Bt[n][k]):
// lane column n fixed, k = kB + idx -> 16 contiguous halves (2x ds_load_b128).
__device__ __forceinline__ v16h load_b_frag_kcontig(const f16* row, int kB) {
  v16h b;
#pragma unroll
  for (int idx = 0; idx < 16; ++idx) b[idx] = row[kB + idx];
  return b;
}

// ---------------------------------------------------------------------------
// 1) fp32 -> f16 weight conversion
// ---------------------------------------------------------------------------
__global__ void cvt_f32_f16(const float* __restrict__ src, f16* __restrict__ dst,
                            int n) {
  int i = blockIdx.x * blockDim.x + threadIdx.x;
  if (i < n) dst[i] = (f16)src[i];
}

// ---------------------------------------------------------------------------
// 2) LayerNorm (row of 768) -> f16
// ---------------------------------------------------------------------------
__launch_bounds__(256)
__global__ void ln_kernel(const float* __restrict__ x,
                          const float* __restrict__ gamma,
                          const float* __restrict__ beta,
                          f16* __restrict__ xn) {
  __shared__ float red[256];
  __shared__ float red2[256];
  const int row = blockIdx.x;
  const int tid = threadIdx.x;
  const float* xr = x + (size_t)row * DIM;
  float v[3], s = 0.f, s2 = 0.f;
#pragma unroll
  for (int i = 0; i < 3; ++i) {
    v[i] = xr[tid + i * 256];
    s += v[i]; s2 += v[i] * v[i];
  }
  red[tid] = s; red2[tid] = s2;
  __syncthreads();
  for (int off = 128; off > 0; off >>= 1) {
    if (tid < off) { red[tid] += red[tid + off]; red2[tid] += red2[tid + off]; }
    __syncthreads();
  }
  const float mean = red[0] * (1.f / DIM);
  const float var  = red2[0] * (1.f / DIM) - mean * mean;   // biased variance
  const float inv  = rsqrtf(var + 1e-5f);
#pragma unroll
  for (int i = 0; i < 3; ++i) {
    const int c = tid + i * 256;
    xn[(size_t)row * DIM + c] = (f16)(((v[i] - mean) * inv) * gamma[c] + beta[c]);
  }
}

// ---------------------------------------------------------------------------
// 3) QKV GEMM (36928x2304 = xn[36928x768] @ w[768x2304]) + RoPE epilogue.
//    Tile 64x64, BK=32, 4 waves. A tile staged via async b128; B tile staged
//    transposed (BsT[n][k]) so fragment reads are contiguous.
// ---------------------------------------------------------------------------
__launch_bounds__(128)
__global__ void qkv_gemm_rope(const f16* __restrict__ xn, const f16* __restrict__ w,
                              const float* __restrict__ fcos,
                              const float* __restrict__ fsin,
                              f16* __restrict__ qh, f16* __restrict__ kh,
                              f16* __restrict__ vh) {
  __shared__ f16  As[64 * 32];      // row-major [m][k]
  __shared__ f16  BsT[64 * 32];     // transposed [n][k]
  __shared__ float Cs[64 * 64];
  const int tid  = threadIdx.x;
  const int lane = tid & 31, wave = tid >> 5;
  const int bm = blockIdx.x % 577;        // M tiles (36928/64)
  const int bn = blockIdx.x / 577;        // N tiles (2304/64)
  const int row0 = bm * 64, col0 = bn * 64;
  const int nloc = wave * 16 + (lane & 15);
  const int kB   = (lane & 16) ? 16 : 0;

  v8f c[4] = {};
  for (int k0 = 0; k0 < DIM; k0 += 32) {
    __syncthreads();
    // A tile: 64 rows x 64B, async copy, 16B granularity (256 chunks)
    for (int j = tid; j < 256; j += 128) {
      const int r = j >> 2, sub = j & 3;
      async_copy_b128(&As[j * 8],
                      (const char*)xn + (((size_t)(row0 + r) * DIM + k0) << 1) + sub * 16);
    }
    // B tile: manual transpose into BsT[n][k]
    {
      const unsigned* srcB = (const unsigned*)w;
      for (int i = tid; i < 32 * 32; i += 128) {
        const int r = i >> 5, cu = i & 31;          // r = k, cu = n-pair
        U32H2 t; t.u = srcB[(size_t)(k0 + r) * (QKVN / 2) + (col0 >> 1) + cu];
        BsT[(2 * cu) * 32 + r]     = t.h[0];
        BsT[(2 * cu + 1) * 32 + r] = t.h[1];
      }
    }
    wait_async0();
    __syncthreads();
    v16h b = load_b_frag_kcontig(&BsT[nloc * 32], kB);
#pragma unroll
    for (int ms = 0; ms < 4; ++ms) {
      v16h a = load_a_frag(As, 32, ms * 16, 0, lane);
      c[ms] = __builtin_amdgcn_wmma_f32_16x16x32_f16(false, a, false, b,
                                                     (short)0, c[ms], false, false);
    }
  }
  __syncthreads();
  {
    const int mofs = (lane & 16) ? 8 : 0;
#pragma unroll
    for (int ms = 0; ms < 4; ++ms)
#pragma unroll
      for (int r = 0; r < 8; ++r)
        Cs[(ms * 16 + r + mofs) * 64 + nloc] = c[ms][r];
  }
  __syncthreads();
  // Epilogue: tile covers exactly one head of one of q/k/v (64 | 768).
  const int region = col0 / DIM;                 // 0=q 1=k 2=v
  const int head   = (col0 % DIM) / DHEAD;
  for (int i = tid; i < 64 * 32; i += 128) {     // element pairs
    const int rr = i >> 5, p = i & 31;
    const int g  = row0 + rr;
    const int b_ = g / NTOK, n_ = g % NTOK;
    const size_t base = ((size_t)(b_ * HEADS + head) * NTOK + n_) * DHEAD;
    const float v0 = Cs[rr * 64 + 2 * p];
    const float v1 = Cs[rr * 64 + 2 * p + 1];
    if (region == 2) {
      vh[base + 2 * p]     = (f16)v0;
      vh[base + 2 * p + 1] = (f16)v1;
    } else {
      float r0 = v0, r1 = v1;
      if (n_ > 0) {                              // RoPE (skip CLS)
        const float ct = fcos[(size_t)(n_ - 1) * (DHEAD / 2) + p];
        const float st = fsin[(size_t)(n_ - 1) * (DHEAD / 2) + p];
        r0 = v0 * ct - v1 * st;
        r1 = v0 * st + v1 * ct;
      }
      if (region == 0) {
        r0 *= 0.125f; r1 *= 0.125f;              // fold 1/sqrt(64) into q
        qh[base + 2 * p] = (f16)r0; qh[base + 2 * p + 1] = (f16)r1;
      } else {
        kh[base + 2 * p] = (f16)r0; kh[base + 2 * p + 1] = (f16)r1;
      }
    }
  }
}

// ---------------------------------------------------------------------------
// 4) Flash attention. Block = (b,h, 64-row q tile); 4 waves.
//    Q/K tiles staged via async b128 (OOB rows zero-filled with DS stores);
//    V staged transposed (VsT[d][kcol]) so the P@V B-fragment is contiguous.
// ---------------------------------------------------------------------------
__launch_bounds__(128)
__global__ void attn_kernel(const f16* __restrict__ qh, const f16* __restrict__ kh,
                            const f16* __restrict__ vh, f16* __restrict__ ao) {
  __shared__ f16  Qs[64 * 64];      // [qrow][d]
  __shared__ f16  Ks[64 * 64];      // [kcol][d]  (S-frag reads d-contiguous)
  __shared__ f16  VsT[64 * 64];     // [d][kcol]  (PV-frag reads k-contiguous)
  __shared__ float Sf[64 * 64];
  __shared__ f16  Ps[64 * 64];
  __shared__ float mS[64], lS[64], aS[64];

  const int tid = threadIdx.x;
  const int lane = tid & 31, wave = tid >> 5;
  const int qt = blockIdx.x % 10;          // ceil(577/64)
  const int bh = blockIdx.x / 10;
  const char* qg = (const char*)qh + ((size_t)bh * NTOK * DHEAD << 1);
  const char* kg = (const char*)kh + ((size_t)bh * NTOK * DHEAD << 1);
  const unsigned* vg = (const unsigned*)(vh + (size_t)bh * NTOK * DHEAD);

  // Q tile: 64 rows x 128B = 512 x b128 chunks
  for (int j = tid; j < 512; j += 128) {
    const int r = j >> 3, sub = j & 7;
    const int qn = qt * 64 + r;
    if (qn < NTOK) {
      async_copy_b128(&Qs[j * 8], qg + (size_t)qn * 128 + sub * 16);
    } else {
      unsigned* z = (unsigned*)&Qs[j * 8];
      z[0] = 0u; z[1] = 0u; z[2] = 0u; z[3] = 0u;
    }
  }
  if (tid < 64) { mS[tid] = NEG_BIG; lS[tid] = 0.f; }
  v8f o[4] = {};
  const int ncol = wave * 16 + (lane & 15);
  const int mofs = (lane & 16) ? 8 : 0;
  const int kB   = (lane & 16) ? 16 : 0;
  wait_async0();
  __syncthreads();

  for (int kc0 = 0; kc0 < NTOK; kc0 += 64) {
    const int kvalid = (NTOK - kc0 < 64) ? (NTOK - kc0) : 64;
    // K tile (async, row-major)
    for (int j = tid; j < 512; j += 128) {
      const int r = j >> 3, sub = j & 7;
      const int kn = kc0 + r;
      if (kn < NTOK) {
        async_copy_b128(&Ks[j * 8], kg + (size_t)kn * 128 + sub * 16);
      } else {
        unsigned* z = (unsigned*)&Ks[j * 8];
        z[0] = 0u; z[1] = 0u; z[2] = 0u; z[3] = 0u;
      }
    }
    // V tile, transposed into VsT[d][kcol]
    for (int i = tid; i < 64 * 32; i += 128) {
      const int r = i >> 5, cu = i & 31;            // r = kcol, cu = d-pair
      const int kn = kc0 + r;
      U32H2 t; t.u = (kn < NTOK) ? vg[(size_t)kn * 32 + cu] : 0u;
      VsT[(2 * cu) * 64 + r]     = t.h[0];
      VsT[(2 * cu + 1) * 64 + r] = t.h[1];
    }
    wait_async0();
    __syncthreads();

    // S = Q @ K^T  (B[k=d][n=kcol] = Ks[kcol][d], d-contiguous per lane)
    v8f s[4] = {};
#pragma unroll
    for (int ks = 0; ks < 2; ++ks) {
      v16h b = load_b_frag_kcontig(&Ks[ncol * 64 + ks * 32], kB);
#pragma unroll
      for (int ms = 0; ms < 4; ++ms) {
        v16h a = load_a_frag(Qs, 64, ms * 16, ks * 32, lane);
        s[ms] = __builtin_amdgcn_wmma_f32_16x16x32_f16(false, a, false, b,
                                                       (short)0, s[ms], false, false);
      }
    }
#pragma unroll
    for (int ms = 0; ms < 4; ++ms)
#pragma unroll
      for (int r = 0; r < 8; ++r)
        Sf[(ms * 16 + r + mofs) * 64 + ncol] = s[ms][r];
    __syncthreads();

    // online softmax, one owner thread per row
    if (tid < 64) {
      const int r = tid;
      const float mo = mS[r];
      float mc = NEG_BIG;
      for (int j = 0; j < kvalid; ++j) mc = fmaxf(mc, Sf[r * 64 + j]);
      const float mn = fmaxf(mo, mc);
      const float al = __expf(mo - mn);
      float sum = 0.f;
      for (int j = 0; j < 64; ++j) {
        const float p = (j < kvalid) ? __expf(Sf[r * 64 + j] - mn) : 0.f;
        Ps[r * 64 + j] = (f16)p;
        sum += p;
      }
      mS[r] = mn;
      lS[r] = lS[r] * al + sum;
      aS[r] = al;
    }
    __syncthreads();

    // rescale running O by per-row alpha, then O += P @ V
#pragma unroll
    for (int ms = 0; ms < 4; ++ms)
#pragma unroll
      for (int r = 0; r < 8; ++r)
        o[ms][r] *= aS[ms * 16 + r + mofs];
#pragma unroll
    for (int ks = 0; ks < 2; ++ks) {
      v16h b = load_b_frag_kcontig(&VsT[ncol * 64 + ks * 32], kB);
#pragma unroll
      for (int ms = 0; ms < 4; ++ms) {
        v16h a = load_a_frag(Ps, 64, ms * 16, ks * 32, lane);
        o[ms] = __builtin_amdgcn_wmma_f32_16x16x32_f16(false, a, false, b,
                                                       (short)0, o[ms], false, false);
      }
    }
    __syncthreads();
  }

  const int b_ = bh / HEADS, head = bh % HEADS;
#pragma unroll
  for (int ms = 0; ms < 4; ++ms)
#pragma unroll
    for (int r = 0; r < 8; ++r) {
      const int m  = ms * 16 + r + mofs;
      const int qn = qt * 64 + m;
      if (qn < NTOK) {
        const float val = o[ms][r] / lS[m];
        ao[((size_t)(b_ * NTOK + qn)) * DIM + head * DHEAD + ncol] = (f16)val;
      }
    }
}

// ---------------------------------------------------------------------------
// 5) Output projection: out = ao[36928x768] @ w_out[768x768] + b_out (fp32)
// ---------------------------------------------------------------------------
__launch_bounds__(128)
__global__ void out_gemm(const f16* __restrict__ ao, const f16* __restrict__ w,
                         const float* __restrict__ bias, float* __restrict__ out) {
  __shared__ f16  As[64 * 32];
  __shared__ f16  BsT[64 * 32];     // transposed [n][k]
  __shared__ float Cs[64 * 64];
  const int tid  = threadIdx.x;
  const int lane = tid & 31, wave = tid >> 5;
  const int bm = blockIdx.x % 577;
  const int bn = blockIdx.x / 577;          // 12 N tiles
  const int row0 = bm * 64, col0 = bn * 64;
  const int nloc = wave * 16 + (lane & 15);
  const int kB   = (lane & 16) ? 16 : 0;

  v8f c[4] = {};
  for (int k0 = 0; k0 < DIM; k0 += 32) {
    __syncthreads();
    for (int j = tid; j < 256; j += 128) {
      const int r = j >> 2, sub = j & 3;
      async_copy_b128(&As[j * 8],
                      (const char*)ao + (((size_t)(row0 + r) * DIM + k0) << 1) + sub * 16);
    }
    {
      const unsigned* srcB = (const unsigned*)w;
      for (int i = tid; i < 32 * 32; i += 128) {
        const int r = i >> 5, cu = i & 31;
        U32H2 t; t.u = srcB[(size_t)(k0 + r) * (DIM / 2) + (col0 >> 1) + cu];
        BsT[(2 * cu) * 32 + r]     = t.h[0];
        BsT[(2 * cu + 1) * 32 + r] = t.h[1];
      }
    }
    wait_async0();
    __syncthreads();
    v16h b = load_b_frag_kcontig(&BsT[nloc * 32], kB);
#pragma unroll
    for (int ms = 0; ms < 4; ++ms) {
      v16h a = load_a_frag(As, 32, ms * 16, 0, lane);
      c[ms] = __builtin_amdgcn_wmma_f32_16x16x32_f16(false, a, false, b,
                                                     (short)0, c[ms], false, false);
    }
  }
  __syncthreads();
  {
    const int mofs = (lane & 16) ? 8 : 0;
#pragma unroll
    for (int ms = 0; ms < 4; ++ms)
#pragma unroll
      for (int r = 0; r < 8; ++r)
        Cs[(ms * 16 + r + mofs) * 64 + nloc] = c[ms][r];
  }
  __syncthreads();
  for (int i = tid; i < 64 * 64; i += 128) {
    const int rr = i >> 6, j = i & 63;
    out[(size_t)(row0 + rr) * DIM + col0 + j] = Cs[rr * 64 + j] + bias[col0 + j];
  }
}

// ---------------------------------------------------------------------------
// Launch
// ---------------------------------------------------------------------------
extern "C" void kernel_launch(void* const* d_in, const int* in_sizes, int n_in,
                              void* d_out, int out_size, void* d_ws, size_t ws_size,
                              hipStream_t stream) {
  const float* x     = (const float*)d_in[0];
  const float* fcos  = (const float*)d_in[1];
  const float* fsin  = (const float*)d_in[2];
  const float* gamma = (const float*)d_in[3];
  const float* beta  = (const float*)d_in[4];
  const float* wqkv  = (const float*)d_in[5];
  const float* wout  = (const float*)d_in[6];
  const float* bout  = (const float*)d_in[7];
  float* out = (float*)d_out;

  char* ws = (char*)d_ws;
  constexpr size_t SZ_ACT = (size_t)ROWS * DIM * sizeof(f16);
  constexpr size_t SZ_WQ  = (size_t)DIM * QKVN * sizeof(f16);
  constexpr size_t SZ_WO  = (size_t)DIM * DIM * sizeof(f16);
  constexpr size_t SZ_QKV = (size_t)BATCH * HEADS * NTOK * DHEAD * sizeof(f16);
  size_t off = 0;
  f16* xn_h   = (f16*)(ws + off); off += SZ_ACT;
  f16* wqkv_h = (f16*)(ws + off); off += SZ_WQ;
  f16* wout_h = (f16*)(ws + off); off += SZ_WO;
  f16* q_h    = (f16*)(ws + off); off += SZ_QKV;
  f16* k_h    = (f16*)(ws + off); off += SZ_QKV;
  f16* v_h    = (f16*)(ws + off); off += SZ_QKV;
  f16* ao_h   = (f16*)(ws + off); off += SZ_ACT;
  (void)ws_size; (void)in_sizes; (void)n_in; (void)out_size;

  {
    const int nq = DIM * QKVN;
    cvt_f32_f16<<<(nq + 255) / 256, 256, 0, stream>>>(wqkv, wqkv_h, nq);
    const int no = DIM * DIM;
    cvt_f32_f16<<<(no + 255) / 256, 256, 0, stream>>>(wout, wout_h, no);
  }
  ln_kernel<<<ROWS, 256, 0, stream>>>(x, gamma, beta, xn_h);
  qkv_gemm_rope<<<577 * (QKVN / 64), 128, 0, stream>>>(xn_h, wqkv_h, fcos, fsin,
                                                       q_h, k_h, v_h);
  attn_kernel<<<BATCH * HEADS * 10, 128, 0, stream>>>(q_h, k_h, v_h, ao_h);
  out_gemm<<<577 * (DIM / 64), 128, 0, stream>>>(ao_h, wout_h, bout, out);
}
